// GeneratedTreeClassifier_50062138802299
// MI455X (gfx1250) — matrix operand order repaired and checked by reading, not compiled, not run on hardware
//
#include <hip/hip_runtime.h>
#include <hip/hip_bf16.h>

// ---------------------------------------------------------------------------
// Soft decision-tree ensemble classifier for MI455X (gfx1250, wave32, WMMA,
// double-buffered async global->LDS DMA staging).
// ---------------------------------------------------------------------------

#define TDEPTH    8
#define NINT      255
#define NLEAF     256
#define NCLS      100
#define NCLS_PAD  112     // 7 WMMA N-tiles of 16 (compute width)
#define NCLS_STG  128     // padded storage rows of leafd for uniform DMA
#define NTREES    16
#define BATCH     8192
#define DFEAT     1024

typedef __attribute__((ext_vector_type(16))) __bf16 v16bf;
typedef __attribute__((ext_vector_type(8)))  float  v8f;

union FragBF { int4 q[2]; v16bf v; };
union AccU   { v8f v; float f[8]; };

__device__ __forceinline__ unsigned short f2bf(float f) {
    unsigned int u = __float_as_uint(f);
    unsigned int r = (u + 0x7FFFu + ((u >> 16) & 1u)) >> 16;  // RNE
    return (unsigned short)r;
}
__device__ __forceinline__ float bf2f(unsigned short h) {
    return __uint_as_float(((unsigned int)h) << 16);
}
__device__ __forceinline__ unsigned pack2(float lo, float hi) {
    return ((unsigned)f2bf(hi) << 16) | f2bf(lo);
}

// --- CDNA5 async global->LDS DMA (16B per lane), tracked by ASYNCcnt -------
__device__ __forceinline__ void async_b128(unsigned lds_byte_off,
                                           unsigned glob_byte_off,
                                           const void* base) {
    asm volatile("global_load_async_to_lds_b128 %0, %1, %2"
                 :: "v"(lds_byte_off), "v"(glob_byte_off), "s"(base)
                 : "memory");
}
__device__ __forceinline__ void wait_async0() {
    asm volatile("s_wait_asynccnt 0x0" ::: "memory");
}
__device__ __forceinline__ void wait_async4() {
    asm volatile("s_wait_asynccnt 0x4" ::: "memory");
}
__device__ __forceinline__ void wait_async6() {
    asm volatile("s_wait_asynccnt 0x6" ::: "memory");
}

// ---------------------------------------------------------------------------
// Kernel: x (f32) -> x_bf (bf16), elementwise. 8 elems / thread.
// ---------------------------------------------------------------------------
__global__ __launch_bounds__(256) void k_cvt_x(
    const float* __restrict__ in, unsigned short* __restrict__ out)
{
    const size_t i = ((size_t)blockIdx.x * 256 + threadIdx.x) * 8;
    const float4 a = reinterpret_cast<const float4*>(in + i)[0];
    const float4 b = reinterpret_cast<const float4*>(in + i)[1];
    uint4 o;
    o.x = pack2(a.x, a.y);
    o.y = pack2(a.z, a.w);
    o.z = pack2(b.x, b.y);
    o.w = pack2(b.z, b.w);
    *reinterpret_cast<uint4*>(out + i) = o;
}

// ---------------------------------------------------------------------------
// Kernel: split_weights (16,255,1024) f32 -> (16,256,1024) bf16, node 255 = 0.
// ---------------------------------------------------------------------------
__global__ __launch_bounds__(256) void k_cvt_w(
    const float* __restrict__ in, unsigned short* __restrict__ out)
{
    const size_t f = ((size_t)blockIdx.x * 256 + threadIdx.x) * 8;
    const int t   = (int)(f >> 18);          // 256*1024
    const int rem = (int)(f & 262143);
    const int n   = rem >> 10;
    uint4 o = make_uint4(0u, 0u, 0u, 0u);
    if (n < NINT) {
        const float* src = in + ((size_t)t * NINT + n) * DFEAT + (rem & 1023);
        const float4 a = reinterpret_cast<const float4*>(src)[0];
        const float4 b = reinterpret_cast<const float4*>(src)[1];
        o.x = pack2(a.x, a.y);
        o.y = pack2(a.z, a.w);
        o.z = pack2(b.x, b.y);
        o.w = pack2(b.z, b.w);
    }
    *reinterpret_cast<uint4*>(out + f) = o;
}

// ---------------------------------------------------------------------------
// Kernel: leaf_dist[t][l][c] = softmax_c(leaf_logits) * softmax_t(tree_w)[t]
// stored bf16, class-major [t][NCLS_STG][NLEAF], rows >= 100 zero.
// ---------------------------------------------------------------------------
__global__ __launch_bounds__(128) void k_prep_leaf(
    const float* __restrict__ leaf_logits,   // (T, 256, 100)
    const float* __restrict__ tree_w,        // (T,)
    unsigned short* __restrict__ leafd)      // (T, 128, 256) bf16
{
    const int t = blockIdx.x >> 8;
    const int l = blockIdx.x & 255;
    const int c = threadIdx.x;               // 0..127

    __shared__ float red[128];

    float wm = -1e30f;
    for (int i = 0; i < NTREES; ++i) wm = fmaxf(wm, tree_w[i]);
    float ws = 0.f;
    for (int i = 0; i < NTREES; ++i) ws += __expf(tree_w[i] - wm);
    const float wt = __expf(tree_w[t] - wm) / ws;

    const float v = (c < NCLS) ? leaf_logits[((size_t)t * NLEAF + l) * NCLS + c]
                               : -1e30f;
    red[c] = v;
    __syncthreads();
    for (int s = 64; s > 0; s >>= 1) {
        if (c < s) red[c] = fmaxf(red[c], red[c + s]);
        __syncthreads();
    }
    const float mx = red[0];
    __syncthreads();

    const float e = (c < NCLS) ? __expf(v - mx) : 0.f;
    red[c] = e;
    __syncthreads();
    for (int s = 64; s > 0; s >>= 1) {
        if (c < s) red[c] += red[c + s];
        __syncthreads();
    }
    const float inv = 1.f / red[0];

    const float o = (c < NCLS) ? (wt * e * inv) : 0.f;
    leafd[((size_t)t * NCLS_STG + c) * NLEAF + l] = f2bf(o);
}

// ---------------------------------------------------------------------------
// Kernel: fused GEMM1 (x @ split_w^T) + bias + sigmoid + soft routing.
// Double-buffered async DMA pipeline: issue chunk kc+1, wait asynccnt<=6
// (the 6 in-flight instructions of chunk kc+1), compute chunk kc.
// ---------------------------------------------------------------------------
__global__ __launch_bounds__(256) void k_split_mu(
    const unsigned short* __restrict__ x_bf,   // (B, D) bf16
    const unsigned short* __restrict__ w_bf,   // (T, 256, D) bf16 (node255=0)
    const float* __restrict__ sb,              // (T, 255)
    unsigned short* __restrict__ mu_out)       // (T, B, 256) bf16
{
    // 64 KB: two 24 KB stage buffers during GEMM, reused for probs after.
    __shared__ alignas(16) unsigned short smem[32768];
    // buffer b (elements): sA at b*12288, sB at b*12288 + 4096

    const int t    = blockIdx.x;
    const int bt   = blockIdx.y;
    const int row0 = bt * 128;
    const int tid  = threadIdx.x;
    const int wave = tid >> 5;
    const int lane = tid & 31;
    const int hi   = lane >> 4;
    const int lo   = lane & 15;

    v8f acc[16];
    for (int i = 0; i < 16; ++i)
        for (int j = 0; j < 8; ++j) acc[i][j] = 0.f;

    // 6 async instructions per chunk per wave: 2 for A (8 KB), 4 for B (16 KB)
    auto issue = [&](int kc, unsigned bufB /*bytes*/) {
        #pragma unroll
        for (int j = 0; j < 2; ++j) {
            const int f = tid + 256 * j;          // 0..511
            const int m = f >> 2;
            const int q = f & 3;
            const unsigned lds  = bufB + (unsigned)((m * 32 + q * 8) * 2);
            const unsigned goff = (unsigned)((((size_t)(row0 + m)) * DFEAT
                                              + kc * 32 + q * 8) * 2);
            async_b128(lds, goff, x_bf);
        }
        #pragma unroll
        for (int j = 0; j < 4; ++j) {
            const int f = tid + 256 * j;          // 0..1023
            const int n = f >> 2;
            const int q = f & 3;
            const unsigned lds  = bufB + 8192u + (unsigned)((n * 32 + q * 8) * 2);
            const unsigned goff = (unsigned)((((size_t)t * 256 + n) * DFEAT
                                              + kc * 32 + q * 8) * 2);
            async_b128(lds, goff, w_bf);
        }
    };

    issue(0, 0u);
    for (int kc = 0; kc < DFEAT / 32; ++kc) {
        const unsigned curE = (unsigned)(kc & 1) * 12288u;   // element offset
        if (kc + 1 < DFEAT / 32) {
            issue(kc + 1, (unsigned)((kc + 1) & 1) * 24576u);
            wait_async6();          // chunk kc fully landed (in-order per wave)
        } else {
            wait_async0();
        }
        __syncthreads();

        const unsigned short* sA = smem + curE;
        const unsigned short* sB = smem + curE + 4096;

        FragBF a;
        {
            const unsigned short* ap = &sA[(wave * 16 + lo) * 32 + hi * 16];
            a.q[0] = *reinterpret_cast<const int4*>(ap);
            a.q[1] = *reinterpret_cast<const int4*>(ap + 8);
        }
        FragBF bb[2];
        {
            const unsigned short* bp = &sB[lo * 32 + hi * 16];
            bb[0].q[0] = *reinterpret_cast<const int4*>(bp);
            bb[0].q[1] = *reinterpret_cast<const int4*>(bp + 8);
        }
        #pragma unroll
        for (int nt = 0; nt < 16; ++nt) {
            const int cur = nt & 1;
            if (nt + 1 < 16) {
                const unsigned short* bp = &sB[((nt + 1) * 16 + lo) * 32 + hi * 16];
                bb[cur ^ 1].q[0] = *reinterpret_cast<const int4*>(bp);
                bb[cur ^ 1].q[1] = *reinterpret_cast<const int4*>(bp + 8);
            }
            acc[nt] = __builtin_amdgcn_wmma_f32_16x16x32_bf16(
                false, a.v, false, bb[cur].v, (short)0, acc[nt], false, false);
        }
        __syncthreads();
    }

    // ---- epilogue: bias + sigmoid -> probs in LDS (reuse full 64 KB)
    unsigned short* sP = smem;               // [128][256] bf16
    #pragma unroll
    for (int nt = 0; nt < 16; ++nt) {
        AccU u; u.v = acc[nt];
        const int n = nt * 16 + lo;
        const float bn = (n < NINT) ? sb[t * NINT + n] : 0.f;
        #pragma unroll
        for (int r = 0; r < 8; ++r) {
            const int m = wave * 16 + r + hi * 8;
            const float logit = u.f[r] + bn;
            const float p = 1.f / (1.f + __expf(-logit));
            sP[m * 256 + n] = f2bf(p);
        }
    }
    __syncthreads();

    // ---- soft routing: 2 threads per row, 128 leaves each
    {
        const int m     = tid >> 1;
        const int lbase = (tid & 1) * 128;
        const size_t ob = ((size_t)t * BATCH + row0 + m) * (size_t)NLEAF;
        #pragma unroll 4
        for (int li = 0; li < 128; ++li) {
            const int l = lbase + li;
            float muv = 1.f;
            int node = 0;
            #pragma unroll
            for (int d = 0; d < TDEPTH; ++d) {
                const int bit = (l >> (7 - d)) & 1;
                const float p = bf2f(sP[m * 256 + node]);
                muv *= bit ? p : (1.f - p);
                node = 2 * node + 1 + bit;
            }
            mu_out[ob + l] = f2bf(muv);
        }
    }
}

// ---------------------------------------------------------------------------
// Kernel: out = sum_t (mu_t @ leafd_t); leafd pre-scaled by tree softmax.
// One flattened 128-chunk pipeline across (tree, kc), double-buffered async
// DMA, 4 uniform async instructions per chunk per wave.
// ---------------------------------------------------------------------------
__global__ __launch_bounds__(256) void k_leaf_mix(
    const unsigned short* __restrict__ mu,      // (T, B, 256) bf16
    const unsigned short* __restrict__ leafd,   // (T, 128, 256) bf16
    float* __restrict__ out)                    // (B, 100)
{
    // two 16 KB stage buffers: buffer b: sA at b*8192, sB at b*8192+4096 (elems)
    __shared__ alignas(16) unsigned short smem[16384];

    const int bt   = blockIdx.x;
    const int tid  = threadIdx.x;
    const int wave = tid >> 5;
    const int lane = tid & 31;
    const int hi   = lane >> 4;
    const int lo   = lane & 15;

    v8f acc[7];
    for (int i = 0; i < 7; ++i)
        for (int j = 0; j < 8; ++j) acc[i][j] = 0.f;

    const int NCHUNK = NTREES * (NLEAF / 32);   // 128

    auto issue = [&](int ck, unsigned bufB /*bytes*/) {
        const int t  = ck >> 3;
        const int kc = ck & 7;
        // A: mu rows (128 x 32 bf16, 512 b128) -> 2 instructions
        #pragma unroll
        for (int j = 0; j < 2; ++j) {
            const int f = tid + 256 * j;
            const int m = f >> 2;
            const int q = f & 3;
            const unsigned lds  = bufB + (unsigned)((m * 32 + q * 8) * 2);
            const unsigned goff = (unsigned)((((size_t)t * BATCH + bt * 128 + m)
                                              * NLEAF + kc * 32 + q * 8) * 2);
            async_b128(lds, goff, mu);
        }
        // B: leafd rows (128 x 32 bf16, 512 b128) -> 2 instructions
        #pragma unroll
        for (int j = 0; j < 2; ++j) {
            const int f = tid + 256 * j;
            const int c = f >> 2;
            const int q = f & 3;
            const unsigned lds  = bufB + 8192u + (unsigned)((c * 32 + q * 8) * 2);
            const unsigned goff = (unsigned)((((size_t)t * NCLS_STG + c)
                                              * NLEAF + kc * 32 + q * 8) * 2);
            async_b128(lds, goff, leafd);
        }
    };

    issue(0, 0u);
    for (int ck = 0; ck < NCHUNK; ++ck) {
        const unsigned curE = (unsigned)(ck & 1) * 8192u;    // element offset
        if (ck + 1 < NCHUNK) {
            issue(ck + 1, (unsigned)((ck + 1) & 1) * 16384u);
            wait_async4();
        } else {
            wait_async0();
        }
        __syncthreads();

        const unsigned short* sA = smem + curE;
        const unsigned short* sB = smem + curE + 4096;

        FragBF a;
        {
            const unsigned short* ap = &sA[(wave * 16 + lo) * 32 + hi * 16];
            a.q[0] = *reinterpret_cast<const int4*>(ap);
            a.q[1] = *reinterpret_cast<const int4*>(ap + 8);
        }
        FragBF bb[2];
        {
            const unsigned short* bp = &sB[lo * 32 + hi * 16];
            bb[0].q[0] = *reinterpret_cast<const int4*>(bp);
            bb[0].q[1] = *reinterpret_cast<const int4*>(bp + 8);
        }
        #pragma unroll
        for (int nt = 0; nt < 7; ++nt) {
            const int cur = nt & 1;
            if (nt + 1 < 7) {
                const unsigned short* bp = &sB[((nt + 1) * 16 + lo) * 32 + hi * 16];
                bb[cur ^ 1].q[0] = *reinterpret_cast<const int4*>(bp);
                bb[cur ^ 1].q[1] = *reinterpret_cast<const int4*>(bp + 8);
            }
            acc[nt] = __builtin_amdgcn_wmma_f32_16x16x32_bf16(
                false, a.v, false, bb[cur].v, (short)0, acc[nt], false, false);
        }
        __syncthreads();
    }

    // ---- epilogue: write f32 output (fully covers d_out, c < 100)
    #pragma unroll
    for (int nt = 0; nt < 7; ++nt) {
        AccU u; u.v = acc[nt];
        const int c = nt * 16 + lo;
        if (c < NCLS) {
            #pragma unroll
            for (int r = 0; r < 8; ++r) {
                const int m = wave * 16 + r + hi * 8;
                out[((size_t)bt * 128 + m) * NCLS + c] = u.f[r];
            }
        }
    }
}

// ---------------------------------------------------------------------------
extern "C" void kernel_launch(void* const* d_in, const int* in_sizes, int n_in,
                              void* d_out, int out_size, void* d_ws, size_t ws_size,
                              hipStream_t stream) {
    const float* x   = (const float*)d_in[0];   // (8192, 1024)
    const float* sw  = (const float*)d_in[1];   // (16, 255, 1024)
    const float* sb  = (const float*)d_in[2];   // (16, 255)
    const float* ll  = (const float*)d_in[3];   // (16, 256, 100)
    const float* tw  = (const float*)d_in[4];   // (16,)
    float* out = (float*)d_out;                 // (8192, 100)

    unsigned short* mu_ws = (unsigned short*)d_ws;                       // 64 MB
    unsigned short* leafd = mu_ws + (size_t)NTREES * BATCH * NLEAF;      // 1 MB
    unsigned short* x_bf  = leafd + (size_t)NTREES * NCLS_STG * NLEAF;   // 16 MB
    unsigned short* w_bf  = x_bf  + (size_t)BATCH * DFEAT;               // 8.4 MB

    // one-time f32 -> bf16 conversion passes
    k_cvt_x<<<(BATCH * DFEAT) / (256 * 8), 256, 0, stream>>>(x, x_bf);
    k_cvt_w<<<(NTREES * 256 * DFEAT) / (256 * 8), 256, 0, stream>>>(sw, w_bf);

    // leaf softmax * tree softmax weight -> bf16, class-major, padded rows
    k_prep_leaf<<<NTREES * NLEAF, 128, 0, stream>>>(ll, tw, leafd);

    // GEMM1 + sigmoid + routing -> mu
    dim3 gA(NTREES, BATCH / 128);
    k_split_mu<<<gA, 256, 0, stream>>>(x_bf, w_bf, sb, mu_ws);

    // GEMM2 accumulated over trees -> output
    k_leaf_mix<<<BATCH / 128, 256, 0, stream>>>(mu_ws, leafd, out);
}
